// SCell_50568944943254
// MI455X (gfx1250) — matrix operand order repaired and checked by reading, not compiled, not run on hardware
//
#include <hip/hip_runtime.h>
#include <math.h>

#define B_    8
#define S_    2048
#define N_    16
#define E_    256
#define H_    256
#define H4_   1024
#define ROWS_ (B_ * S_)   // 16384

typedef __attribute__((ext_vector_type(16))) __bf16         v16bf;
typedef __attribute__((ext_vector_type(8)))  float          v8f;
typedef __attribute__((ext_vector_type(16))) unsigned short us16;
typedef __attribute__((ext_vector_type(8)))  unsigned short us8;

__device__ __forceinline__ unsigned short f2bf(float f) {
  unsigned u = __float_as_uint(f);
  u += 0x7FFFu + ((u >> 16) & 1u);          // round-to-nearest-even
  return (unsigned short)(u >> 16);
}

__device__ __forceinline__ v8f v8f_zero() {
  return (v8f){0.f, 0.f, 0.f, 0.f, 0.f, 0.f, 0.f, 0.f};
}

// Low 32 bits of a generic pointer to LDS == LDS byte address (aperture rule).
__device__ __forceinline__ unsigned lds_addr32(const void* p) {
  return (unsigned)(unsigned long long)p;
}

// CDNA5 async copy: 16B memory -> LDS, tracked by ASYNCcnt (no VGPR staging).
__device__ __forceinline__ void async_copy_b128(unsigned lds_off, const void* gptr) {
  asm volatile("global_load_async_to_lds_b128 %0, %1, off"
               :: "v"(lds_off), "v"((unsigned long long)gptr) : "memory");
}

__device__ __forceinline__ void wait_async0() {
  asm volatile("s_wait_asynccnt 0x0" ::: "memory");
}

union U16x16 { us16 u; struct { us8 lo; us8 hi; } p; };

// A fragment (16x32 bf16, row-major source, ld elements per row).
// Lane l holds row (row0 + (l&15)); K range [k0+16*(l>=16), +16)
__device__ __forceinline__ v16bf load_a_frag(const unsigned short* mat, int ld,
                                             int row0, int k0) {
  int lane = threadIdx.x & 31;
  const unsigned short* r =
      mat + (size_t)(row0 + (lane & 15)) * ld + k0 + ((lane >> 4) << 4);
  U16x16 t;
  t.p.lo = *(const us8*)(r);
  t.p.hi = *(const us8*)(r + 8);
  return __builtin_bit_cast(v16bf, t.u);
}

// B fragment (32x16 bf16) from a pre-transposed weight WT[Ncol][K].
// Lane l holds column (n0 + (l&15)), K range [k0 + 16*(l>=16), +16): one 32B load.
__device__ __forceinline__ v16bf load_b_frag(const unsigned short* wt, int ldk,
                                             int n0, int k0) {
  int lane = threadIdx.x & 31;
  const unsigned short* r =
      wt + (size_t)(n0 + (lane & 15)) * ldk + k0 + ((lane >> 4) << 4);
  us16 t = *(const us16*)(r);
  return __builtin_bit_cast(v16bf, t);
}

__device__ __forceinline__ v8f wmma_bf16(v16bf a, v16bf b, v8f c) {
  return __builtin_amdgcn_wmma_f32_16x16x32_bf16(false, a, false, b,
                                                 (short)0, c, false, false);
}

__device__ __forceinline__ float sigm(float x) {
  return 1.f / (1.f + __expf(-x));
}

// ---------------- conversion kernels ----------------

__global__ void cvt_kernel(const float* __restrict__ src,
                           unsigned short* __restrict__ dst, int n) {
  int i = blockIdx.x * 256 + threadIdx.x;
  if (i < n) dst[i] = f2bf(src[i]);
}

// dst[n*K + k] = bf16(src[k*Ncols + n])  (transpose + convert)
__global__ void cvt_t_kernel(const float* __restrict__ src,
                             unsigned short* __restrict__ dst, int K, int Ncols) {
  int i = blockIdx.x * 256 + threadIdx.x;
  if (i < K * Ncols) {
    int k = i % K;
    int n = i / K;
    dst[i] = f2bf(src[(size_t)k * Ncols + n]);
  }
}

// gveca[b][k] = (g @ V_a + Vb_a)[b][k] ; gvec[b][k4] = (g @ V + Vb)[b][k4]
__global__ void gvec_kernel(const float* __restrict__ g,
                            const float* __restrict__ Va, const float* __restrict__ Vba,
                            const float* __restrict__ V,  const float* __restrict__ Vb,
                            float* __restrict__ gveca, float* __restrict__ gvec) {
  int i = blockIdx.x * 256 + threadIdx.x;
  if (i < B_ * H_) {
    int b = i >> 8, k = i & (H_ - 1);
    float s = Vba[k];
    for (int hh = 0; hh < H_; ++hh) s += g[b * H_ + hh] * Va[hh * H_ + k];
    gveca[i] = s;
  } else if (i < B_ * H_ + B_ * H4_) {
    int j = i - B_ * H_;
    int b = j >> 10, k = j & (H4_ - 1);
    float s = Vb[k];
    for (int hh = 0; hh < H_; ++hh) s += g[b * H_ + hh] * V[hh * H4_ + k];
    gvec[j] = s;
  }
}

// ---------------- base = h@Wh_a + x@U_a + gveca ----------------
// grid.x = 512 blocks of 32 rows; block = 128 (4 waves).
// Wave w owns N-tiles w*4..w*4+3 and BOTH 16-row M-tiles (B-fragment reuse x2).
// A staged in LDS via async global->LDS copies; k-loop software-pipelined.

__global__ void __launch_bounds__(128)
base_kernel(const unsigned short* __restrict__ hb, const unsigned short* __restrict__ xb,
            const unsigned short* __restrict__ WhTa, const unsigned short* __restrict__ UTa,
            const float* __restrict__ gveca, float* __restrict__ basef) {
  __shared__ __attribute__((aligned(32))) unsigned short At[32 * H_];  // 16 KB

  int tid  = threadIdx.x;
  int wave = tid >> 5, lane = tid & 31;
  int row0 = blockIdx.x * 32;
  int nt0  = wave * 4;

  v8f acc[2][4];
#pragma unroll
  for (int mi = 0; mi < 2; ++mi)
#pragma unroll
    for (int j = 0; j < 4; ++j) acc[mi][j] = v8f_zero();

  const unsigned short* Asrc[2] = {hb, xb};
  const unsigned short* Wsrc[2] = {WhTa, UTa};
  unsigned atl = lds_addr32(At);

#pragma unroll
  for (int p = 0; p < 2; ++p) {
    // async stage of 32x256 bf16 A block (16 KB): 8x b128 per thread, no VGPRs
    {
      const uint4* src = (const uint4*)(Asrc[p] + (size_t)row0 * H_);
#pragma unroll
      for (int q = 0; q < 8; ++q)
        async_copy_b128(atl + (unsigned)(tid + q * 128) * 16, src + tid + q * 128);
      wait_async0();
    }
    __syncthreads();

    const unsigned short* WT = Wsrc[p];
    v16bf a0 = load_a_frag(At, H_, 0, 0);
    v16bf a1 = load_a_frag(At, H_, 16, 0);
    v16bf b0 = load_b_frag(WT, H_, (nt0 + 0) * 16, 0);
    v16bf b1 = load_b_frag(WT, H_, (nt0 + 1) * 16, 0);
    v16bf b2 = load_b_frag(WT, H_, (nt0 + 2) * 16, 0);
    v16bf b3 = load_b_frag(WT, H_, (nt0 + 3) * 16, 0);

#pragma unroll
    for (int kk = 0; kk < 8; ++kk) {
      v16bf na0 = a0, na1 = a1, nb0 = b0, nb1 = b1, nb2 = b2, nb3 = b3;
      if (kk < 7) {                      // compile-time after full unroll
        int k0 = (kk + 1) * 32;
        na0 = load_a_frag(At, H_, 0, k0);
        na1 = load_a_frag(At, H_, 16, k0);
        nb0 = load_b_frag(WT, H_, (nt0 + 0) * 16, k0);
        nb1 = load_b_frag(WT, H_, (nt0 + 1) * 16, k0);
        nb2 = load_b_frag(WT, H_, (nt0 + 2) * 16, k0);
        nb3 = load_b_frag(WT, H_, (nt0 + 3) * 16, k0);
      }
      acc[0][0] = wmma_bf16(a0, b0, acc[0][0]);
      acc[1][0] = wmma_bf16(a1, b0, acc[1][0]);
      acc[0][1] = wmma_bf16(a0, b1, acc[0][1]);
      acc[1][1] = wmma_bf16(a1, b1, acc[1][1]);
      acc[0][2] = wmma_bf16(a0, b2, acc[0][2]);
      acc[1][2] = wmma_bf16(a1, b2, acc[1][2]);
      acc[0][3] = wmma_bf16(a0, b3, acc[0][3]);
      acc[1][3] = wmma_bf16(a1, b3, acc[1][3]);
      a0 = na0; a1 = na1; b0 = nb0; b1 = nb1; b2 = nb2; b3 = nb3;
    }
    __syncthreads();   // before At is overwritten by next phase
  }

  int mbase = (lane >> 4) * 8;
  int b = row0 >> 11;                 // constant: 32-row block never crosses b
#pragma unroll
  for (int mi = 0; mi < 2; ++mi) {
#pragma unroll
    for (int j = 0; j < 4; ++j) {
      int col = (nt0 + j) * 16 + (lane & 15);
      float gadd = gveca[b * H_ + col];
#pragma unroll
      for (int v = 0; v < 8; ++v) {
        int row = row0 + mi * 16 + mbase + v;
        basef[(size_t)row * H_ + col] = acc[mi][j][v] + gadd;
      }
    }
  }
}

// ---------------- neighbor attention ----------------
// one block (128 thr = 4 waves) per (b,s); neighbors = gathered(16x256) @ Wn_a
// Neighbor gather is done with async global->LDS copies (predicated on idx!=0).

__global__ void __launch_bounds__(128)
attn_kernel(const unsigned short* __restrict__ hb, const unsigned short* __restrict__ WnTa,
            const float* __restrict__ basef,
            const int* __restrict__ nidx, const int* __restrict__ nmask,
            const float* __restrict__ u_w, const float* __restrict__ u_b,
            unsigned short* __restrict__ hnb) {
  __shared__ __attribute__((aligned(32))) unsigned short gA[16 * H_];  // 8 KB
  __shared__ float neigh[16 * H_];                                     // 16 KB
  __shared__ float red[128];
  __shared__ float logits[N_];
  __shared__ float score[N_];

  int tid = threadIdx.x;
  int bs  = blockIdx.x;           // b*S_ + s
  int b   = bs >> 11;

  // gather: 8 threads per neighbor row, 64B per thread (4x async b128 or zeros)
  {
    int rowg = tid >> 3;
    int part = tid & 7;
    int iv   = nidx[bs * N_ + rowg];
    unsigned dstl = lds_addr32(gA) + (unsigned)(rowg * H_ + part * 32) * 2;
    if (iv != 0) {
      const uint4* src =
          (const uint4*)(hb + ((size_t)b * S_ + (iv - 1)) * H_) + part * 4;
#pragma unroll
      for (int q = 0; q < 4; ++q)
        async_copy_b128(dstl + (unsigned)q * 16, src + q);
    } else {
      uint4* dst = (uint4*)(gA + rowg * H_);
#pragma unroll
      for (int q = 0; q < 4; ++q)
        dst[part * 4 + q] = make_uint4(0u, 0u, 0u, 0u);
    }
    wait_async0();
  }
  __syncthreads();

  int wave = tid >> 5, lane = tid & 31;
  int nt0  = wave * 4;
  v8f acc[4];
#pragma unroll
  for (int j = 0; j < 4; ++j) acc[j] = v8f_zero();

  // software-pipelined GEMM: A from LDS, B prefetched from global
  v16bf a  = load_a_frag(gA, H_, 0, 0);
  v16bf b0 = load_b_frag(WnTa, H_, (nt0 + 0) * 16, 0);
  v16bf b1 = load_b_frag(WnTa, H_, (nt0 + 1) * 16, 0);
  v16bf b2 = load_b_frag(WnTa, H_, (nt0 + 2) * 16, 0);
  v16bf b3 = load_b_frag(WnTa, H_, (nt0 + 3) * 16, 0);
#pragma unroll
  for (int kk = 0; kk < 8; ++kk) {
    v16bf na = a, nb0 = b0, nb1 = b1, nb2 = b2, nb3 = b3;
    if (kk < 7) {
      int k0 = (kk + 1) * 32;
      na  = load_a_frag(gA, H_, 0, k0);
      nb0 = load_b_frag(WnTa, H_, (nt0 + 0) * 16, k0);
      nb1 = load_b_frag(WnTa, H_, (nt0 + 1) * 16, k0);
      nb2 = load_b_frag(WnTa, H_, (nt0 + 2) * 16, k0);
      nb3 = load_b_frag(WnTa, H_, (nt0 + 3) * 16, k0);
    }
    acc[0] = wmma_bf16(a, b0, acc[0]);
    acc[1] = wmma_bf16(a, b1, acc[1]);
    acc[2] = wmma_bf16(a, b2, acc[2]);
    acc[3] = wmma_bf16(a, b3, acc[3]);
    a = na; b0 = nb0; b1 = nb1; b2 = nb2; b3 = nb3;
  }

  // apply neighbor mask, spill masked neighbors (16 x 256 f32) to LDS
  int mbase = (lane >> 4) * 8;
  float mf[8];
#pragma unroll
  for (int v = 0; v < 8; ++v) mf[v] = (float)nmask[bs * N_ + mbase + v];
#pragma unroll
  for (int j = 0; j < 4; ++j) {
    int col = (nt0 + j) * 16 + (lane & 15);
#pragma unroll
    for (int v = 0; v < 8; ++v) {
      neigh[(mbase + v) * H_ + col] = acc[j][v] * mf[v];
    }
  }
  __syncthreads();

  // logits[m] = dot(base + neigh[m], u_w) + u_b
  {
    int rowg = tid >> 3, part = tid & 7;
    float p = 0.f;
    for (int q = 0; q < 32; ++q) {
      int cidx = part * 32 + q;
      p += (neigh[rowg * H_ + cidx] + basef[(size_t)bs * H_ + cidx]) * u_w[cidx];
    }
    red[tid] = p;
  }
  __syncthreads();
  if (tid < N_) {
    float s = u_b[0];
    for (int q = 0; q < 8; ++q) s += red[tid * 8 + q];
    float m = (float)nmask[bs * N_ + tid];
    logits[tid] = s * (1.0f - m) * 1e-25f;
  }
  __syncthreads();
  if (tid < N_) {
    float mx = -1e30f;
    for (int q = 0; q < N_; ++q) mx = fmaxf(mx, logits[q]);
    float sum = 0.f;
    for (int q = 0; q < N_; ++q) sum += __expf(logits[q] - mx);
    score[tid] = __expf(logits[tid] - mx) / sum;
  }
  __syncthreads();

  // hn[col] = sum_n score[n] * neigh[n][col]
  for (int col = tid; col < H_; col += 128) {
    float s = 0.f;
#pragma unroll
    for (int n = 0; n < N_; ++n) s += score[n] * neigh[n * H_ + col];
    hnb[(size_t)bs * H_ + col] = f2bf(s);
  }
}

// ---------------- gates + LSTM cell ----------------
// grid (512, 4); block 128 (4 waves). Wave owns the same 16 columns of all four
// gates (N-offsets g*256 + c0) for 32 rows, so i/f/o/u stay in registers.

__global__ void __launch_bounds__(128)
gates_kernel(const unsigned short* __restrict__ hb, const unsigned short* __restrict__ xb,
             const unsigned short* __restrict__ hnb,
             const unsigned short* __restrict__ WhT, const unsigned short* __restrict__ UT,
             const unsigned short* __restrict__ WnT,
             const float* __restrict__ gvec, const float* __restrict__ c_in,
             float* __restrict__ out) {
  __shared__ __attribute__((aligned(32))) unsigned short At[32 * H_];  // 16 KB

  int tid  = threadIdx.x;
  int wave = tid >> 5, lane = tid & 31;
  int row0 = blockIdx.x * 32;
  int c0   = (blockIdx.y * 4 + wave) * 16;   // 0..240

  v8f acc[2][4];
#pragma unroll
  for (int mi = 0; mi < 2; ++mi)
#pragma unroll
    for (int g = 0; g < 4; ++g) acc[mi][g] = v8f_zero();

  const unsigned short* As[3] = {hb, xb, hnb};
  const unsigned short* Ws[3] = {WhT, UT, WnT};
  unsigned atl = lds_addr32(At);

#pragma unroll
  for (int m3 = 0; m3 < 3; ++m3) {
    {
      const uint4* src = (const uint4*)(As[m3] + (size_t)row0 * H_);
#pragma unroll
      for (int q = 0; q < 8; ++q)
        async_copy_b128(atl + (unsigned)(tid + q * 128) * 16, src + tid + q * 128);
      wait_async0();
    }
    __syncthreads();

    const unsigned short* WT = Ws[m3];
    v16bf a0 = load_a_frag(At, H_, 0, 0);
    v16bf a1 = load_a_frag(At, H_, 16, 0);
    v16bf b0 = load_b_frag(WT, H_, 0 * 256 + c0, 0);
    v16bf b1 = load_b_frag(WT, H_, 1 * 256 + c0, 0);
    v16bf b2 = load_b_frag(WT, H_, 2 * 256 + c0, 0);
    v16bf b3 = load_b_frag(WT, H_, 3 * 256 + c0, 0);

#pragma unroll
    for (int kk = 0; kk < 8; ++kk) {
      v16bf na0 = a0, na1 = a1, nb0 = b0, nb1 = b1, nb2 = b2, nb3 = b3;
      if (kk < 7) {
        int k0 = (kk + 1) * 32;
        na0 = load_a_frag(At, H_, 0, k0);
        na1 = load_a_frag(At, H_, 16, k0);
        nb0 = load_b_frag(WT, H_, 0 * 256 + c0, k0);
        nb1 = load_b_frag(WT, H_, 1 * 256 + c0, k0);
        nb2 = load_b_frag(WT, H_, 2 * 256 + c0, k0);
        nb3 = load_b_frag(WT, H_, 3 * 256 + c0, k0);
      }
      acc[0][0] = wmma_bf16(a0, b0, acc[0][0]);
      acc[1][0] = wmma_bf16(a1, b0, acc[1][0]);
      acc[0][1] = wmma_bf16(a0, b1, acc[0][1]);
      acc[1][1] = wmma_bf16(a1, b1, acc[1][1]);
      acc[0][2] = wmma_bf16(a0, b2, acc[0][2]);
      acc[1][2] = wmma_bf16(a1, b2, acc[1][2]);
      acc[0][3] = wmma_bf16(a0, b3, acc[0][3]);
      acc[1][3] = wmma_bf16(a1, b3, acc[1][3]);
      a0 = na0; a1 = na1; b0 = nb0; b1 = nb1; b2 = nb2; b3 = nb3;
    }
    __syncthreads();
  }

  int mbase = (lane >> 4) * 8;
  int col   = c0 + (lane & 15);
  int b     = row0 >> 11;
  const float* gv = gvec + b * H4_;
  float gvi = gv[col], gvf = gv[256 + col], gvo = gv[512 + col], gvu = gv[768 + col];

#pragma unroll
  for (int mi = 0; mi < 2; ++mi) {
#pragma unroll
    for (int v = 0; v < 8; ++v) {
      int row = row0 + mi * 16 + mbase + v;
      float gi = acc[mi][0][v] + gvi;
      float gf = acc[mi][1][v] + gvf;
      float go = acc[mi][2][v] + gvo;
      float gu = acc[mi][3][v] + gvu;
      float cin = c_in[(size_t)row * H_ + col];
      float nc = sigm(gf) * cin + sigm(gi) * tanhf(gu);
      float nh = sigm(go) * tanhf(nc);
      out[(size_t)row * H_ + col] = nh;                          // new_h
      out[(size_t)ROWS_ * H_ + (size_t)row * H_ + col] = nc;     // new_c
    }
  }
}

// ---------------- host launcher ----------------

extern "C" void kernel_launch(void* const* d_in, const int* in_sizes, int n_in,
                              void* d_out, int out_size, void* d_ws, size_t ws_size,
                              hipStream_t stream) {
  const float* x    = (const float*)d_in[0];
  const float* h    = (const float*)d_in[1];
  const float* c    = (const float*)d_in[2];
  const float* g    = (const float*)d_in[3];
  const int*   nidx = (const int*)d_in[4];
  const int*   nmask= (const int*)d_in[5];
  const float* Wh_a = (const float*)d_in[6];
  const float* Wn_a = (const float*)d_in[7];
  const float* U_a  = (const float*)d_in[8];
  const float* u_w  = (const float*)d_in[9];
  const float* u_b  = (const float*)d_in[10];
  const float* V_a  = (const float*)d_in[11];
  const float* Vb_a = (const float*)d_in[12];
  const float* Wh   = (const float*)d_in[13];
  const float* Wn   = (const float*)d_in[14];
  const float* U    = (const float*)d_in[15];
  const float* V    = (const float*)d_in[16];
  const float* Vb   = (const float*)d_in[17];

  char* ws = (char*)d_ws;
  size_t off = 0;
  auto alloc = [&](size_t bytes) -> void* {
    void* p = ws + off;
    off += (bytes + 255) & ~(size_t)255;
    return p;
  };

  unsigned short* hb    = (unsigned short*)alloc((size_t)ROWS_ * H_ * 2);   // 8 MiB
  unsigned short* xb    = (unsigned short*)alloc((size_t)ROWS_ * E_ * 2);   // 8 MiB
  unsigned short* hnb   = (unsigned short*)alloc((size_t)ROWS_ * H_ * 2);   // 8 MiB
  float*          basef = (float*)alloc((size_t)ROWS_ * H_ * 4);            // 16 MiB
  unsigned short* WhTa  = (unsigned short*)alloc((size_t)H_ * H_ * 2);
  unsigned short* WnTa  = (unsigned short*)alloc((size_t)H_ * H_ * 2);
  unsigned short* UTa   = (unsigned short*)alloc((size_t)E_ * H_ * 2);
  unsigned short* WhT   = (unsigned short*)alloc((size_t)H_ * H4_ * 2);
  unsigned short* UT    = (unsigned short*)alloc((size_t)E_ * H4_ * 2);
  unsigned short* WnT   = (unsigned short*)alloc((size_t)H_ * H4_ * 2);
  float*          gveca = (float*)alloc((size_t)B_ * H_ * 4);
  float*          gvec  = (float*)alloc((size_t)B_ * H4_ * 4);

  // 1) bf16 conversions (activations + transposed weights)
  int nact = ROWS_ * H_;
  cvt_kernel<<<(nact + 255) / 256, 256, 0, stream>>>(h, hb, nact);
  cvt_kernel<<<(nact + 255) / 256, 256, 0, stream>>>(x, xb, nact);
  cvt_t_kernel<<<(H_ * H_ + 255) / 256, 256, 0, stream>>>(Wh_a, WhTa, H_, H_);
  cvt_t_kernel<<<(H_ * H_ + 255) / 256, 256, 0, stream>>>(Wn_a, WnTa, H_, H_);
  cvt_t_kernel<<<(E_ * H_ + 255) / 256, 256, 0, stream>>>(U_a, UTa, E_, H_);
  cvt_t_kernel<<<(H_ * H4_ + 255) / 256, 256, 0, stream>>>(Wh, WhT, H_, H4_);
  cvt_t_kernel<<<(E_ * H4_ + 255) / 256, 256, 0, stream>>>(U, UT, E_, H4_);
  cvt_t_kernel<<<(H_ * H4_ + 255) / 256, 256, 0, stream>>>(Wn, WnT, H_, H4_);

  // 2) g-dependent bias rows (tiny)
  gvec_kernel<<<(B_ * (H_ + H4_) + 255) / 256, 256, 0, stream>>>(
      g, V_a, Vb_a, V, Vb, gveca, gvec);

  // 3) base = h@Wh_a + x@U_a + gveca
  base_kernel<<<dim3(ROWS_ / 32), dim3(128), 0, stream>>>(hb, xb, WhTa, UTa,
                                                          gveca, basef);

  // 4) neighbor attention -> hn (bf16)
  attn_kernel<<<dim3(ROWS_), dim3(128), 0, stream>>>(hb, WnTa, basef, nidx,
                                                     nmask, u_w, u_b, hnb);

  // 5) gates + LSTM cell -> new_h, new_c
  gates_kernel<<<dim3(ROWS_ / 32, 4), dim3(128), 0, stream>>>(
      hb, xb, hnb, WhT, UT, WnT, gvec, c, (float*)d_out);
}
